// SelfAttention_76974403879346
// MI455X (gfx1250) — compile-verified, matrix operand order checked by reading, not run on hardware
//
#include <hip/hip_runtime.h>
#include <hip/hip_bf16.h>

typedef __attribute__((ext_vector_type(16))) _Float16 v16h;
typedef __attribute__((ext_vector_type(8)))  _Float16 v8h;
typedef __attribute__((ext_vector_type(8)))  float    v8f;

#define SEQ 2048
#define DMODEL 2048
#define NKV 512           // 8 kv heads * 64
#define HD 64

// ---------------------------------------------------------------------------
// Fragment loader: builds a 16x32 f16 WMMA A-fragment (or B-fragment, where
// "rows" are the columns of B, i.e. Wt / K / Vt rows) from row-major memory.
// Per ISA 16-bit A layout: lanes 0-15 hold K 0..7 (v0-3) and 16..23 (v4-7);
// lanes 16-31 hold K 8..15 and 24..31.  Two contiguous b128 loads per lane.
// ---------------------------------------------------------------------------
__device__ __forceinline__ v16h load_frag16(const _Float16* __restrict__ base,
                                            int row0, int lda, int k0, int lane) {
  const int lrow = lane & 15;
  const int halfid = lane >> 4;
  const _Float16* p = base + (size_t)(row0 + lrow) * (size_t)lda + k0 + halfid * 8;
  union { v16h v; v8h h2[2]; } u;
  u.h2[0] = *(const v8h*)(p);
  u.h2[1] = *(const v8h*)(p + 16);
  return u.v;
}

__device__ __forceinline__ v8f wmma32(v16h a, v16h b, v8f c) {
  return __builtin_amdgcn_wmma_f32_16x16x32_f16(
      /*neg_a=*/false, a, /*neg_b=*/false, b,
      /*c_mod=*/(short)0, c, /*reuse_a=*/false, /*reuse_b=*/false);
}

__device__ __forceinline__ float rmax16(float x) {
#pragma unroll
  for (int off = 1; off < 16; off <<= 1) x = fmaxf(x, __shfl_xor(x, off, 32));
  return x;
}
__device__ __forceinline__ float rsum16(float x) {
#pragma unroll
  for (int off = 1; off < 16; off <<= 1) x += __shfl_xor(x, off, 32);
  return x;
}

// ---------------------------------------------------------------------------
// Elementwise converters
// ---------------------------------------------------------------------------
__global__ void cvt_f32_f16(const float* __restrict__ in, _Float16* __restrict__ out, int n) {
  int i = blockIdx.x * blockDim.x + threadIdx.x;
  if (i < n) out[i] = (_Float16)in[i];
}

// out[n*K + k] = (f16) in[k*N + n]   (weight transpose for contiguous B frags)
__global__ void transpose_cvt(const float* __restrict__ in, _Float16* __restrict__ out,
                              int K, int N) {
  int i = blockIdx.x * blockDim.x + threadIdx.x;
  if (i < K * N) {
    int k = i / N, n = i - k * N;
    out[(size_t)n * K + k] = (_Float16)in[i];
  }
}

// ---------------------------------------------------------------------------
// RoPE on K (in place, [S][512]) and V (read [S][512], write transposed
// Vt[feature][S] so P*V B-fragments are contiguous).
// ---------------------------------------------------------------------------
__global__ void rope_kv(const float* __restrict__ tc, const float* __restrict__ ts,
                        _Float16* __restrict__ Kh,
                        const _Float16* __restrict__ Vp,
                        _Float16* __restrict__ Vt) {
  int i = blockIdx.x * blockDim.x + threadIdx.x;   // SEQ * 256 pairs
  if (i >= SEQ * (NKV / 2)) return;
  int s  = i >> 8;
  int pr = i & 255;
  int f0 = pr << 1;
  int fi = pr & 31;                                // freq index within head (hd/2 = 32)
  float c  = tc[s * 32 + fi];
  float sn = ts[s * 32 + fi];
  {
    _Float16* kp = Kh + (size_t)s * NKV + f0;
    float x0 = (float)kp[0], x1 = (float)kp[1];
    kp[0] = (_Float16)(x0 * c - x1 * sn);
    kp[1] = (_Float16)(x0 * sn + x1 * c);
  }
  {
    const _Float16* vp = Vp + (size_t)s * NKV + f0;
    float x0 = (float)vp[0], x1 = (float)vp[1];
    Vt[(size_t)f0 * SEQ + s]       = (_Float16)(x0 * c - x1 * sn);
    Vt[(size_t)(f0 + 1) * SEQ + s] = (_Float16)(x0 * sn + x1 * c);
  }
}

// ---------------------------------------------------------------------------
// Generic WMMA GEMM: C[M][N] = A[M][K] * Wt[N][K]^T, f32 accumulate.
// Block = 256 threads = 8 waves arranged 2(M) x 4(N); block tile 64 x 256.
// Wave tile 32 x 64: two A fragments share each B fragment -> 8 WMMAs per
// 12 b128 loads (~21 FLOP/byte from L2, double the naive 16x64 tile).
// out_f32 != 0 -> store float (final projection straight into d_out).
// ---------------------------------------------------------------------------
__global__ __launch_bounds__(256)
void gemm_f16_wmma(const _Float16* __restrict__ A,
                   const _Float16* __restrict__ Wt,
                   void* __restrict__ Cout,
                   int K, int ldc, int out_f32) {
  const int lane = threadIdx.x & 31;
  const int wave = threadIdx.x >> 5;
  const int wm = wave >> 2;            // 0..1
  const int wn = wave & 3;             // 0..3
  const int m0 = blockIdx.y * 64 + wm * 32;
  const int n0 = blockIdx.x * 256 + wn * 64;
  const int lrow = lane & 15;
  const int halfid = lane >> 4;

  v8f acc[2][4] = {};

  for (int k0 = 0; k0 < K; k0 += 32) {
    // prefetch next A blocks into cache hierarchy (global_prefetch_b8)
    if (k0 + 64 < K) {
      __builtin_prefetch(A + (size_t)(m0 + lrow) * K + k0 + 64, 0, 1);
      __builtin_prefetch(A + (size_t)(m0 + 16 + lrow) * K + k0 + 64, 0, 1);
    }
    v16h a0 = load_frag16(A, m0,      K, k0, lane);
    v16h a1 = load_frag16(A, m0 + 16, K, k0, lane);
#pragma unroll
    for (int t = 0; t < 4; ++t) {
      v16h b = load_frag16(Wt, n0 + t * 16, K, k0, lane);
      acc[0][t] = wmma32(a0, b, acc[0][t]);
      acc[1][t] = wmma32(a1, b, acc[1][t]);
    }
  }

#pragma unroll
  for (int s = 0; s < 2; ++s) {
#pragma unroll
    for (int t = 0; t < 4; ++t) {
#pragma unroll
      for (int v = 0; v < 8; ++v) {
        int row = m0 + s * 16 + v + 8 * halfid;
        int col = n0 + t * 16 + lrow;
        if (out_f32) ((float*)Cout)[(size_t)row * ldc + col] = acc[s][t][v];
        else ((_Float16*)Cout)[(size_t)row * ldc + col] = (_Float16)acc[s][t][v];
      }
    }
  }
}

// ---------------------------------------------------------------------------
// Flash attention: one wave per (head, 16-query tile). Online softmax over
// 32-key blocks; 4 score WMMAs + 4 PV WMMAs per block; causal mask.
// Q: [S][2048] f16 (head offset h*64), K: [S][512] f16 (roped),
// Vt: [512][S] f16 (roped, transposed), Y: [S][2048] f16.
// ---------------------------------------------------------------------------
__global__ __launch_bounds__(128)
void attn_wmma(const _Float16* __restrict__ Qh,
               const _Float16* __restrict__ Kh,
               const _Float16* __restrict__ Vt,
               _Float16* __restrict__ Yh) {
  __shared__ __align__(16) _Float16 pbuf[4][16][40];   // per-wave P tile, padded

  const int lane = threadIdx.x & 31;
  const int w = threadIdx.x >> 5;
  const int wid = blockIdx.x * 4 + w;        // 0..4095
  const int h  = wid >> 7;                   // head 0..31
  const int q0 = (wid & 127) << 4;           // query tile base
  const int kvh = h >> 2;                    // GQA: 4 q-heads per kv-head
  const int lrow = lane & 15;
  const int halfid = lane >> 4;

  // Q fragments for this tile (reused across all key blocks)
  const _Float16* Qbase = Qh + h * HD;
  v16h qa0 = load_frag16(Qbase, q0, DMODEL, 0, lane);
  v16h qa1 = load_frag16(Qbase, q0, DMODEL, 32, lane);

  const _Float16* Khead = Kh + kvh * HD;
  const _Float16* Vhead = Vt + (size_t)kvh * HD * SEQ;

  v8f acc[4] = {};
  float m_r[8], l_r[8];
#pragma unroll
  for (int v = 0; v < 8; ++v) { m_r[v] = -1e30f; l_r[v] = 0.0f; }

  const int nb = (q0 + 16 + 31) >> 5;        // 32-key blocks (causal bound)
  for (int b = 0; b < nb; ++b) {
    const int n0 = b << 5;

    // ---- scores: S = Q * K^T over head_dim 64 (2 x K32 chain per subtile)
    v8f s0 = {}, s1 = {};
    {
      v16h kb = load_frag16(Khead, n0, NKV, 0, lane);
      s0 = wmma32(qa0, kb, s0);
      kb = load_frag16(Khead, n0, NKV, 32, lane);
      s0 = wmma32(qa1, kb, s0);
      kb = load_frag16(Khead, n0 + 16, NKV, 0, lane);
      s1 = wmma32(qa0, kb, s1);
      kb = load_frag16(Khead, n0 + 16, NKV, 32, lane);
      s1 = wmma32(qa1, kb, s1);
    }

    const bool need_mask = (n0 + 31 > q0);
    float p0[8], p1[8];
#pragma unroll
    for (int v = 0; v < 8; ++v) {
      const int qrow = q0 + v + 8 * halfid;
      float a0 = s0[v] * 0.125f;             // hd^-0.5, hd = 64
      float a1 = s1[v] * 0.125f;
      if (need_mask) {
        if (n0 + lrow > qrow)      a0 = -1e30f;
        if (n0 + 16 + lrow > qrow) a1 = -1e30f;
      }
      float mb   = rmax16(fmaxf(a0, a1));
      float mnew = fmaxf(m_r[v], mb);
      float cf   = __expf(m_r[v] - mnew);
      m_r[v] = mnew;
      float e0 = __expf(a0 - mnew);
      float e1 = __expf(a1 - mnew);
      p0[v] = e0; p1[v] = e1;
      l_r[v] = l_r[v] * cf + rsum16(e0 + e1);
#pragma unroll
      for (int t = 0; t < 4; ++t) acc[t][v] *= cf;
    }

    // ---- C-layout -> A-layout relayout of P through per-wave LDS tile
#pragma unroll
    for (int v = 0; v < 8; ++v) {
      pbuf[w][v + 8 * halfid][lrow]      = (_Float16)p0[v];
      pbuf[w][v + 8 * halfid][16 + lrow] = (_Float16)p1[v];
    }
    v16h pa;
    {
      const _Float16* pp = &pbuf[w][lrow][halfid * 8];
      union { v16h v; v8h h2[2]; } u;
      u.h2[0] = *(const v8h*)(pp);
      u.h2[1] = *(const v8h*)(pp + 16);
      pa = u.v;
    }

    // ---- acc += P * V   (B frags from transposed V rows, contiguous)
#pragma unroll
    for (int t = 0; t < 4; ++t) {
      v16h vb = load_frag16(Vhead, t * 16, SEQ, n0, lane);
      acc[t] = wmma32(pa, vb, acc[t]);
    }
  }

  // ---- epilogue: normalize and store y[q][h*64 + d] as f16
#pragma unroll
  for (int t = 0; t < 4; ++t) {
#pragma unroll
    for (int v = 0; v < 8; ++v) {
      int row = q0 + v + 8 * halfid;
      int col = h * HD + t * 16 + lrow;
      Yh[(size_t)row * DMODEL + col] = (_Float16)(acc[t][v] * (1.0f / l_r[v]));
    }
  }
}

// ---------------------------------------------------------------------------
// Host-side orchestration (all on `stream`, graph-capture safe).
// ---------------------------------------------------------------------------
extern "C" void kernel_launch(void* const* d_in, const int* in_sizes, int n_in,
                              void* d_out, int out_size, void* d_ws, size_t ws_size,
                              hipStream_t stream) {
  const float* x  = (const float*)d_in[0];
  const float* tc = (const float*)d_in[1];
  const float* ts = (const float*)d_in[2];
  const float* wq = (const float*)d_in[3];
  const float* wk = (const float*)d_in[4];
  const float* wv = (const float*)d_in[5];
  const float* wo = (const float*)d_in[6];

  char* ws = (char*)d_ws;
  // workspace layout (bytes)
  _Float16* xh  = (_Float16*)(ws + 0);           //  8 MB  x f16            [2048][2048]
  _Float16* wqT = (_Float16*)(ws + (8u  << 20)); //  8 MB  wq^T f16        [2048][2048]
  _Float16* wkT = (_Float16*)(ws + (16u << 20)); //  2 MB  wk^T f16        [512][2048]
  _Float16* wvT = (_Float16*)(ws + (18u << 20)); //  2 MB  wv^T f16        [512][2048]
  _Float16* woT = (_Float16*)(ws + (20u << 20)); //  8 MB  wo^T f16        [2048][2048]
  _Float16* Qh  = (_Float16*)(ws + (28u << 20)); //  8 MB  Q f16           [2048][2048]
  _Float16* Kh  = (_Float16*)(ws + (36u << 20)); //  2 MB  K f16 (roped)   [2048][512]
  _Float16* Vp  = (_Float16*)(ws + (38u << 20)); //  2 MB  V f16 pre-rope  [2048][512]
  _Float16* Vt  = (_Float16*)(ws + (40u << 20)); //  2 MB  V f16 roped^T   [512][2048]
  _Float16* Yh  = (_Float16*)(ws + (42u << 20)); //  8 MB  attn out f16    [2048][2048]

  const int TPB = 256;
  // 1) stage activations + transposed weights in f16
  cvt_f32_f16<<<(DMODEL * SEQ + TPB - 1) / TPB, TPB, 0, stream>>>(x, xh, DMODEL * SEQ);
  transpose_cvt<<<(DMODEL * DMODEL + TPB - 1) / TPB, TPB, 0, stream>>>(wq, wqT, DMODEL, DMODEL);
  transpose_cvt<<<(DMODEL * NKV   + TPB - 1) / TPB, TPB, 0, stream>>>(wk, wkT, DMODEL, NKV);
  transpose_cvt<<<(DMODEL * NKV   + TPB - 1) / TPB, TPB, 0, stream>>>(wv, wvT, DMODEL, NKV);
  transpose_cvt<<<(DMODEL * DMODEL + TPB - 1) / TPB, TPB, 0, stream>>>(wo, woT, DMODEL, DMODEL);

  // 2) projections (WMMA GEMMs)
  gemm_f16_wmma<<<dim3(DMODEL / 256, SEQ / 64), 256, 0, stream>>>(xh, wqT, Qh, DMODEL, DMODEL, 0);
  gemm_f16_wmma<<<dim3(NKV    / 256, SEQ / 64), 256, 0, stream>>>(xh, wkT, Kh, DMODEL, NKV, 0);
  gemm_f16_wmma<<<dim3(NKV    / 256, SEQ / 64), 256, 0, stream>>>(xh, wvT, Vp, DMODEL, NKV, 0);

  // 3) RoPE on K (in place) and V (into transposed layout)
  rope_kv<<<(SEQ * (NKV / 2) + TPB - 1) / TPB, TPB, 0, stream>>>(tc, ts, Kh, Vp, Vt);

  // 4) causal flash attention (WMMA)
  attn_wmma<<<dim3((32 * (SEQ / 16)) / 4), 128, 0, stream>>>(Qh, Kh, Vt, Yh);

  // 5) output projection, f32 straight into d_out
  gemm_f16_wmma<<<dim3(DMODEL / 256, SEQ / 64), 256, 0, stream>>>(Yh, woT, d_out, DMODEL, DMODEL, 1);
}
